// GRUDImputer_4234837753878
// MI455X (gfx1250) — compile-verified, matrix operand order checked by reading, not run on hardware
//
#include <hip/hip_runtime.h>

#define Bsz 1024
#define Tsz 1024
#define Hd  128
#define Cd  6
#define OUTHALF ((long)Bsz * Tsz * Cd)

typedef __attribute__((ext_vector_type(16))) _Float16 v16h;
typedef __attribute__((ext_vector_type(8)))  float    v8f;

union Frag {
    v16h h;
    uint4 u[2];
};

__device__ __forceinline__ v8f wmma_f16(v16h a, v16h b, v8f c) {
    return __builtin_amdgcn_wmma_f32_16x16x32_f16(
        /*neg_a=*/false, a, /*neg_b=*/false, b,
        /*c_mod=*/(short)0, c, /*reuse_a=*/false, /*reuse_b=*/false);
}

// Load a pre-packed B fragment from global (fragment-ordered blob, 32B/lane, coalesced).
__device__ __forceinline__ v16h load_frag_g(const unsigned short* base, int tile_chunk, int lane) {
    const uint4* p = (const uint4*)(base + ((long)tile_chunk * 32 + lane) * 16);
    Frag f;
    f.u[0] = p[0];
    f.u[1] = p[1];
    return f.h;
}

// Load a pre-packed B fragment from LDS (same fragment order).
__device__ __forceinline__ v16h load_frag_lds(const _Float16* base, int tile_chunk, int lane) {
    const uint4* p = (const uint4*)(base + ((tile_chunk * 32 + lane) << 4));
    Frag f;
    f.u[0] = p[0];
    f.u[1] = p[1];
    return f.h;
}

// Load an A fragment (16xK row-major f16 in LDS) per ISA 16-bit A-matrix layout:
// lanes 0-15:  K = 32c + {0..7, 16..23};  lanes 16-31: K = 32c + {8..15, 24..31}
__device__ __forceinline__ v16h load_a_lds(const _Float16* lds, int K, int c, int lane) {
    int row = lane & 15;
    int off = row * K + 32 * c + ((lane < 16) ? 0 : 8);
    Frag f;
    f.u[0] = *(const uint4*)(lds + off);
    f.u[1] = *(const uint4*)(lds + off + 16);
    return f.h;
}

__device__ __forceinline__ float sigmoidf_(float x) { return 1.0f / (1.0f + __expf(-x)); }
__device__ __forceinline__ float softplusf_(float x) { return (x > 20.0f) ? x : log1pf(__expf(x)); }

// ---------------- Weight fragment prep ----------------
// One thread per f16 element. Packs W[N][K] (row-major f32, zero-padded to
// 16*NT x 32*NC) into WMMA B-fragment order.
__global__ void grud_prep(const float* __restrict__ W, unsigned short* __restrict__ dst,
                          int Kvalid, int Nvalid, int NC, long total) {
    long idx = (long)blockIdx.x * blockDim.x + threadIdx.x;
    if (idx >= total) return;
    int  q  = (int)(idx & 15);
    int  l  = (int)((idx >> 4) & 31);
    long tc = idx >> 9;
    int  c  = (int)(tc % NC);
    long j  = tc / NC;
    int  n  = (int)(16 * j + (l & 15));
    int  k  = 32 * c + ((l < 16) ? ((q < 8) ? q : 8 + q) : ((q < 8) ? 8 + q : 16 + q));
    float val = (n < Nvalid && k < Kvalid) ? W[(long)n * Kvalid + k] : 0.0f;
    _Float16 hv = (_Float16)val;
    dst[idx] = __builtin_bit_cast(unsigned short, hv);
}

// ws blob layout (fragment index -> matrix), 1 KB per fragment:
//   [  0.. 95] w_hh   (24 tiles x 4 chunks)   -> LDS
//   [ 96..111] wu1    ( 4 tiles x 4 chunks)   -> LDS
//   [112..135] w_ih   (24 tiles x 1 chunk)    -> LDS
//   [136..139] w2     ( 1 tile  x 4 chunks)   -> LDS
//   [140..141] wu2    ( 1 tile  x 2 chunks)   -> LDS
//   [142..173] w1     ( 8 tiles x 4 chunks)   -> global (L2-hot), per step
#define NFRAG_LDS 142

// ---------------- Main recurrent kernel ----------------
// One workgroup (4 waves, 128 threads) per 16-row batch tile. The per-step
// column work is split across the 4 waves:
//   gates : wave w owns hidden tiles j = {2w, 2w+1} (r, z, n and the h update)
//   t1    : wave w owns tiles {2w, 2w+1};  u1: wave w owns tile w
//   pred  : wave 0;  unc: wave 1;  input/x_hat path: wave 3 (lanes 0..15)
// h is exchanged through LDS each step (needed for A fragments anyway).
__global__ __launch_bounds__(128) void grud_main(
    const float* __restrict__ x,
    const float* __restrict__ x_mean,
    const float* __restrict__ dxw, const float* __restrict__ dxb,
    const float* __restrict__ dhw, const float* __restrict__ dhb,
    const float* __restrict__ b_ih, const float* __restrict__ b_hh,
    const float* __restrict__ b1,  const float* __restrict__ b2,
    const float* __restrict__ bu1, const float* __restrict__ bu2,
    const unsigned short* __restrict__ wfrag,
    float* __restrict__ out)
{
    __shared__ __align__(16) _Float16 sh_w[NFRAG_LDS * 512];   // 142 KB
    __shared__ __align__(16) _Float16 sh_h [16 * 128];
    __shared__ __align__(16) _Float16 sh_t1[16 * 128];
    __shared__ __align__(16) _Float16 sh_u1[16 * 64];
    __shared__ __align__(16) _Float16 sh_gin[16 * 32];
    __shared__ float sh_dt[16];

    const int tid  = threadIdx.x;        // 0..127
    const int lane = tid & 31;
    const int wu   = __builtin_amdgcn_readfirstlane(tid >> 5);  // uniform wave id 0..3
    const int b0   = blockIdx.x * 16;    // batch tile base
    const int nn   = lane & 15;          // column within a 16-wide N tile
    const int mrow = (lane >> 4) * 8;    // D-layout row base (0 or 8)

    // ---- Stage LDS-resident weights (one contiguous range, coalesced b128 copies) ----
    {
        const uint4* src = (const uint4*)wfrag;
        uint4* dw = (uint4*)sh_w;
        for (int i = tid; i < NFRAG_LDS * 64; i += 128) dw[i] = src[i];
    }

    // Per-lane bias / decay constants for the tiles this wave owns
    float Br[2], Bz[2], Bin[2], Bhn[2], B1v[2], DW8[2], DB8[2];
#pragma unroll
    for (int jj = 0; jj < 2; ++jj) {
        int n = 16 * (2 * wu + jj) + nn;
        Br[jj]  = b_ih[n]       + b_hh[n];
        Bz[jj]  = b_ih[n + 128] + b_hh[n + 128];
        Bin[jj] = b_ih[n + 256];
        Bhn[jj] = b_hh[n + 256];
        B1v[jj] = b1[n];
        DW8[jj] = dhw[n];
        DB8[jj] = dhb[n];
    }
    float Bu1v = bu1[16 * wu + nn];
    float B2v  = (nn < Cd) ? b2[nn]  : 0.0f;
    float Bu2v = (nn < Cd) ? bu2[nn] : 0.0f;

    // Input-path constants & state (used by wave 3, lanes 0..15)
    float xm[6], dw6[6], db6[6];
#pragma unroll
    for (int c = 0; c < 6; ++c) { xm[c] = x_mean[c]; dw6[c] = dxw[c]; db6[c] = dxb[c]; }
    float running[6] = {0, 0, 0, 0, 0, 0};
    float xlast[6]   = {0, 0, 0, 0, 0, 0};

    // Hidden state (this wave's 2 tiles) in f32 C/D layout:
    // element (m = mrow+v, n = 16*(2wu+jj)+nn)
    float h[2][8];
#pragma unroll
    for (int jj = 0; jj < 2; ++jj)
#pragma unroll
        for (int v = 0; v < 8; ++v) h[jj][v] = 0.0f;

    // Zero gru_in pad region once (cols 12..31 stay zero forever)
    for (int i = tid; i < 16 * 32; i += 128) sh_gin[i] = (_Float16)0.0f;
    __syncthreads();

#pragma unroll 1
    for (int t = 0; t < Tsz; ++t) {
        // Opaque zero offsets: stop LICM from hoisting per-step weight-fragment
        // loads out of the loop (which would pin hundreds of VGPRs and spill).
        unsigned wo = 0;
        asm volatile("" : "+s"(wo));
        const _Float16* wl = sh_w + wo;
        unsigned long go = 0;
        asm volatile("" : "+s"(go));
        const unsigned short* F_w1 = wfrag + (long)NFRAG_LDS * 512 + go;

        // ---- 1. Per-row input processing (wave 3, lanes 0..15; one batch row each) ----
        if (wu == 3) {
            if (lane < 16) {
                const float* xr = x + (((long)(b0 + lane)) * Tsz + t) * 13;
                int tn = (t + 1 < Tsz) ? t + 1 : t;
                __builtin_prefetch(x + (((long)(b0 + lane)) * Tsz + tn) * 13, 0, 1);
                float xv[6], mv[6];
#pragma unroll
                for (int c = 0; c < 6; ++c) { xv[c] = xr[c]; mv[c] = xr[6 + c]; }
                float dtv = xr[12];
                sh_dt[lane] = dtv;
#pragma unroll
                for (int c = 0; c < 6; ++c) {
                    bool obs = mv[c] > 0.5f;
                    running[c] = obs ? 0.0f : running[c] + dtv;
                    float gx = __expf(-fmaxf(running[c] * dw6[c] + db6[c], 0.0f));
                    xlast[c] = obs ? xv[c] : xlast[c];
                    float xh = mv[c] * xv[c] + (1.0f - mv[c]) * (gx * xlast[c] + (1.0f - gx) * xm[c]);
                    sh_gin[lane * 32 + c]     = (_Float16)xh;
                    sh_gin[lane * 32 + 6 + c] = (_Float16)mv[c];
                }
            }
        }
        __syncthreads();

        // ---- 2. Decay this wave's h tiles by gamma_h, publish decayed h (f16) ----
        float dtm[8];
#pragma unroll
        for (int v = 0; v < 8; ++v) dtm[v] = sh_dt[mrow + v];
#pragma unroll
        for (int jj = 0; jj < 2; ++jj) {
            const int j = 2 * wu + jj;
#pragma unroll
            for (int v = 0; v < 8; ++v) {
                float gh = __expf(-fmaxf(dtm[v] * DW8[jj] + DB8[jj], 0.0f));
                h[jj][v] *= gh;
                sh_h[(mrow + v) * 128 + 16 * j + nn] = (_Float16)h[jj][v];
            }
        }
        __syncthreads();

        // ---- 3. Load A fragments (full h + gru_in) ----
        v16h Ah[4];
#pragma unroll
        for (int c = 0; c < 4; ++c) Ah[c] = load_a_lds(sh_h, 128, c, lane);
        v16h Agi = load_a_lds(sh_gin, 32, 0, lane);

        // ---- 4. Gates for this wave's 2 hidden tiles; update h in place ----
#pragma unroll
        for (int jj = 0; jj < 2; ++jj) {
            const int j = 2 * wu + jj;
            v8f accr  = {Br[jj], Br[jj], Br[jj], Br[jj], Br[jj], Br[jj], Br[jj], Br[jj]};
            v8f accz  = {Bz[jj], Bz[jj], Bz[jj], Bz[jj], Bz[jj], Bz[jj], Bz[jj], Bz[jj]};
            v8f accin = {Bin[jj], Bin[jj], Bin[jj], Bin[jj], Bin[jj], Bin[jj], Bin[jj], Bin[jj]};
            v8f acchn = {Bhn[jj], Bhn[jj], Bhn[jj], Bhn[jj], Bhn[jj], Bhn[jj], Bhn[jj], Bhn[jj]};
            accr  = wmma_f16(Agi, load_frag_lds(wl, 112 + j,      lane), accr);
            accz  = wmma_f16(Agi, load_frag_lds(wl, 112 + j + 8,  lane), accz);
            accin = wmma_f16(Agi, load_frag_lds(wl, 112 + j + 16, lane), accin);
#pragma unroll
            for (int c = 0; c < 4; ++c) {
                accr  = wmma_f16(Ah[c], load_frag_lds(wl, (j)      * 4 + c, lane), accr);
                accz  = wmma_f16(Ah[c], load_frag_lds(wl, (j + 8)  * 4 + c, lane), accz);
                acchn = wmma_f16(Ah[c], load_frag_lds(wl, (j + 16) * 4 + c, lane), acchn);
            }
#pragma unroll
            for (int v = 0; v < 8; ++v) {
                float r  = sigmoidf_(accr[v]);
                float zz = sigmoidf_(accz[v]);
                float nv = tanhf(accin[v] + r * acchn[v]);
                h[jj][v] = (1.0f - zz) * nv + zz * h[jj][v];
            }
        }
        // Publish new h (f16) for the head matmuls
#pragma unroll
        for (int jj = 0; jj < 2; ++jj)
#pragma unroll
            for (int v = 0; v < 8; ++v)
                sh_h[(mrow + v) * 128 + 16 * (2 * wu + jj) + nn] = (_Float16)h[jj][v];
        __syncthreads();

        v16h Ahn[4];
#pragma unroll
        for (int c = 0; c < 4; ++c) Ahn[c] = load_a_lds(sh_h, 128, c, lane);

        // ---- 5. t1 tiles {2wu, 2wu+1} (w1 from global/L2), u1 tile {wu} ----
#pragma unroll
        for (int jj = 0; jj < 2; ++jj) {
            const int j = 2 * wu + jj;
            v8f acc = {B1v[jj], B1v[jj], B1v[jj], B1v[jj], B1v[jj], B1v[jj], B1v[jj], B1v[jj]};
#pragma unroll
            for (int c = 0; c < 4; ++c)
                acc = wmma_f16(Ahn[c], load_frag_g(F_w1, j * 4 + c, lane), acc);
#pragma unroll
            for (int v = 0; v < 8; ++v)
                sh_t1[(mrow + v) * 128 + 16 * j + nn] = (_Float16)fmaxf(acc[v], 0.0f);
        }
        {
            v8f acc = {Bu1v, Bu1v, Bu1v, Bu1v, Bu1v, Bu1v, Bu1v, Bu1v};
#pragma unroll
            for (int c = 0; c < 4; ++c)
                acc = wmma_f16(Ahn[c], load_frag_lds(wl, 96 + wu * 4 + c, lane), acc);
#pragma unroll
            for (int v = 0; v < 8; ++v)
                sh_u1[(mrow + v) * 64 + 16 * wu + nn] = (_Float16)fmaxf(acc[v], 0.0f);
        }
        __syncthreads();

        // ---- 6. pred (wave 0) / unc (wave 1) + output stores ----
        if (wu == 0) {
            v8f accp = {B2v, B2v, B2v, B2v, B2v, B2v, B2v, B2v};
#pragma unroll
            for (int c = 0; c < 4; ++c)
                accp = wmma_f16(load_a_lds(sh_t1, 128, c, lane),
                                load_frag_lds(wl, 136 + c, lane), accp);
            if (nn < Cd) {
#pragma unroll
                for (int v = 0; v < 8; ++v) {
                    long row = ((long)(b0 + mrow + v)) * Tsz + t;
                    out[row * Cd + nn] = accp[v];
                }
            }
        } else if (wu == 1) {
            v8f accu = {Bu2v, Bu2v, Bu2v, Bu2v, Bu2v, Bu2v, Bu2v, Bu2v};
#pragma unroll
            for (int c = 0; c < 2; ++c)
                accu = wmma_f16(load_a_lds(sh_u1, 64, c, lane),
                                load_frag_lds(wl, 140 + c, lane), accu);
            if (nn < Cd) {
#pragma unroll
                for (int v = 0; v < 8; ++v) {
                    long row = ((long)(b0 + mrow + v)) * Tsz + t;
                    out[OUTHALF + row * Cd + nn] = softplusf_(accu[v]);
                }
            }
        }
        __syncthreads();
    }
}

extern "C" void kernel_launch(void* const* d_in, const int* in_sizes, int n_in,
                              void* d_out, int out_size, void* d_ws, size_t ws_size,
                              hipStream_t stream) {
    const float* x      = (const float*)d_in[0];
    const float* x_mean = (const float*)d_in[1];
    const float* dxw    = (const float*)d_in[2];
    const float* dxb    = (const float*)d_in[3];
    const float* dhw    = (const float*)d_in[4];
    const float* dhb    = (const float*)d_in[5];
    const float* w_ih   = (const float*)d_in[6];
    const float* w_hh   = (const float*)d_in[7];
    const float* b_ih   = (const float*)d_in[8];
    const float* b_hh   = (const float*)d_in[9];
    const float* w1     = (const float*)d_in[10];
    const float* b1     = (const float*)d_in[11];
    const float* w2     = (const float*)d_in[12];
    const float* b2     = (const float*)d_in[13];
    const float* wu1    = (const float*)d_in[14];
    const float* bu1    = (const float*)d_in[15];
    const float* wu2    = (const float*)d_in[16];
    const float* bu2    = (const float*)d_in[17];

    unsigned short* ws = (unsigned short*)d_ws;

    auto prep = [&](const float* W, long off_halves, int Kv, int Nv, int NT, int NC) {
        long total = (long)NT * NC * 512;
        int blocks = (int)((total + 255) / 256);
        grud_prep<<<blocks, 256, 0, stream>>>(W, ws + off_halves, Kv, Nv, NC, total);
    };
    // LDS-resident range (one contiguous copy in grud_main):
    prep(w_hh, 0,     128, 384, 24, 4);   // frags   0.. 95
    prep(wu1,  49152, 128, 64,  4,  4);   // frags  96..111
    prep(w_ih, 57344, 12,  384, 24, 1);   // frags 112..135
    prep(w2,   69632, 128, 6,   1,  4);   // frags 136..139
    prep(wu2,  71680, 64,  6,   1,  2);   // frags 140..141
    // Global-resident (read per step from L2):
    prep(w1,   72704, 128, 128, 8,  4);   // frags 142..173

    grud_main<<<Bsz / 16, 128, 0, stream>>>(
        x, x_mean, dxw, dxb, dhw, dhb, b_ih, b_hh, b1, b2, bu1, bu2,
        (const unsigned short*)ws, (float*)d_out);
}
